// LINEAR_32298154066288
// MI455X (gfx1250) — matrix operand order repaired and checked by reading, not compiled, not run on hardware
//
#include <hip/hip_runtime.h>

// Problem constants
#define NB  64
#define TSTEPS 1024
#define II  88
#define HH  1024
#define GWG 16            // persistent recurrence workgroups (one WGP each)
#define JCH (HH / GWG)    // 64 output columns per workgroup

typedef __bf16 bf16_t;
typedef __attribute__((ext_vector_type(16))) __bf16        v16bf;
typedef __attribute__((ext_vector_type(8)))  float         v8f;
typedef __attribute__((ext_vector_type(4)))  float         f32x4;
typedef __attribute__((ext_vector_type(4)))  unsigned int  u32x4;
typedef __attribute__((ext_vector_type(8)))  int           i32x8;
typedef __attribute__((ext_vector_type(4)))  int           i32x4;

struct pair16 { u32x4 lo, hi; };   // 32 bytes -> one v16bf fragment

// ---------------------------------------------------------------------------
// Kernel 0: zero barrier words in workspace (must be deterministic per call)
// ---------------------------------------------------------------------------
__global__ void init_ws(int* p) {
    if (threadIdx.x < 16) p[threadIdx.x] = 0;
}

// ---------------------------------------------------------------------------
// Kernel 1: ih[r][h] = x[r]·W_ih[h] + b_ih[h], r = n*T + t   (bf16 WMMA)
// Block = 256 thr (8 waves). Out tile per block: 128 rows x 64 cols.
// Wave w: m-tile = w, n-tiles 0..3. K = 88 -> 3 k-blocks of 32 (last padded).
// ---------------------------------------------------------------------------
__global__ __launch_bounds__(256) void input_gemm(
    const float* __restrict__ x, const float* __restrict__ W_ih,
    const float* __restrict__ b_ih, float* __restrict__ ih)
{
    const int tid   = threadIdx.x;
    const int lane  = tid & 31;
    const int wv    = tid >> 5;
    const int lhalf = lane >> 4;
    const int l15   = lane & 15;

    const int rowA    = blockIdx.y * 128 + wv * 16 + l15;  // A row (M = lane%16)
    const int colBase = blockIdx.x * 64;

    v8f c[4] = {v8f{}, v8f{}, v8f{}, v8f{}};

    #pragma unroll
    for (int kb = 0; kb < 3; ++kb) {
        // ---- A fragment: lane<16 holds K k0..k0+7 & k0+16..k0+23 ----
        v16bf a;
        {
            const float* xr = x + (size_t)rowA * II;
            const int k0 = kb * 32 + lhalf * 8;
            #pragma unroll
            for (int ch = 0; ch < 2; ++ch) {
                const int ks = k0 + ch * 16;
                if (ks < II) {
                    f32x4 f0 = *(const f32x4*)(xr + ks);
                    f32x4 f1 = *(const f32x4*)(xr + ks + 4);
                    #pragma unroll
                    for (int e = 0; e < 4; ++e) {
                        a[ch * 8 + e]     = (bf16_t)f0[e];
                        a[ch * 8 + 4 + e] = (bf16_t)f1[e];
                    }
                } else {
                    #pragma unroll
                    for (int e = 0; e < 8; ++e) a[ch * 8 + e] = (bf16_t)0.0f;
                }
            }
        }
        // ---- B fragments: lane holds col n=lane%16, K k0..k0+7 & k0+16..k0+23 ----
        #pragma unroll
        for (int nt = 0; nt < 4; ++nt) {
            v16bf b;
            const int j  = colBase + nt * 16 + l15;
            const int ks = kb * 32 + lhalf * 8;
            const float* wr = W_ih + (size_t)j * II;
            #pragma unroll
            for (int hc = 0; hc < 2; ++hc) {
                const int kk = ks + hc * 16;
                if (kk < II) {
                    f32x4 f0 = *(const f32x4*)(wr + kk);
                    f32x4 f1 = *(const f32x4*)(wr + kk + 4);
                    #pragma unroll
                    for (int e = 0; e < 4; ++e) {
                        b[hc * 8 + e]     = (bf16_t)f0[e];
                        b[hc * 8 + 4 + e] = (bf16_t)f1[e];
                    }
                } else {
                    #pragma unroll
                    for (int e = 0; e < 8; ++e) b[hc * 8 + e] = (bf16_t)0.0f;
                }
            }
            c[nt] = __builtin_amdgcn_wmma_f32_16x16x32_bf16(
                        false, a, false, b, (short)0, c[nt], false, false);
        }
    }

    // epilogue: +bias, store fp32
    #pragma unroll
    for (int nt = 0; nt < 4; ++nt) {
        const int col = colBase + nt * 16 + l15;
        const float bb = b_ih[col];
        #pragma unroll
        for (int r = 0; r < 8; ++r) {
            const int row = blockIdx.y * 128 + wv * 16 + r + lhalf * 8;
            ih[(size_t)row * HH + col] = c[nt][r] + bb;
        }
    }
}

// ---------------------------------------------------------------------------
// Global barrier across GWG persistent workgroups
// ---------------------------------------------------------------------------
__device__ __forceinline__ void grid_barrier(int* cnt, int* gen) {
    __threadfence();          // release this thread's stores to device scope
    __syncthreads();
    if (threadIdx.x == 0) {
        int g = __hip_atomic_load(gen, __ATOMIC_RELAXED, __HIP_MEMORY_SCOPE_AGENT);
        int prev = __hip_atomic_fetch_add(cnt, 1, __ATOMIC_ACQ_REL, __HIP_MEMORY_SCOPE_AGENT);
        if (prev == GWG - 1) {
            __hip_atomic_store(cnt, 0, __ATOMIC_RELAXED, __HIP_MEMORY_SCOPE_AGENT);
            __hip_atomic_fetch_add(gen, 1, __ATOMIC_RELEASE, __HIP_MEMORY_SCOPE_AGENT);
        } else {
            while (__hip_atomic_load(gen, __ATOMIC_ACQUIRE, __HIP_MEMORY_SCOPE_AGENT) == g) {
                __builtin_amdgcn_s_sleep(2);
            }
        }
    }
    __syncthreads();
    __threadfence();          // acquire side
}

// ---------------------------------------------------------------------------
// Kernel 2: persistent recurrence. GWG=16 blocks x 256 thr (8 waves).
// Block b owns output columns j0..j0+63. W_hh slice lives in LDS in WMMA
// B-fragment order (bf16). Each step: TDM-load h_{t-1} (bf16, 64x1024) into
// LDS (padded: row stride 520 dwords), 64x64 = 16 WMMA tiles over K=1024.
// ---------------------------------------------------------------------------
#define WFRAG_BYTES  (4 * 32 * 1024)        // 4 ni-tiles * 32 kb * 32 lanes * 32B = 128 KB
#define AROW_BYTES   2080                   // 1024 bf16 + TDM pad (2*16B)
#define ASTAGE_BYTES (NB * AROW_BYTES)      // 133120 B
#define SMEM_BYTES   (WFRAG_BYTES + ASTAGE_BYTES)

__global__ __launch_bounds__(256) void rnn_recur(
    const float* __restrict__ initial, const float* __restrict__ W_hh,
    const float* __restrict__ ihbuf,   float* __restrict__ hid,
    bf16_t* __restrict__ hbuf0, bf16_t* __restrict__ hbuf1,
    int* cnt, int* gen)
{
    __shared__ __align__(128) unsigned char smem[SMEM_BYTES];
    unsigned char* wfrag  = smem;
    unsigned char* astage = smem + WFRAG_BYTES;

    const int tid   = threadIdx.x;
    const int lane  = tid & 31;
    const int wv    = tid >> 5;
    const int lhalf = lane >> 4;
    const int l15   = lane & 15;
    const int j0    = blockIdx.x * JCH;

    // ---- one-time: stage W_hh slice (rows j0..j0+63) as bf16 B-fragments ----
    // fragment f = ni*32+kb; lane l: row j0+ni*16+(l&15), K = kb*32+(l>>4)*16 .. +15
    for (int f = wv * 16; f < wv * 16 + 16; ++f) {
        const int ni = f >> 5, kb = f & 31;
        const int j  = j0 + ni * 16 + l15;
        const int ks = kb * 32 + lhalf * 16;
        const float* wr = W_hh + (size_t)j * HH + ks;
        bf16_t* dst = (bf16_t*)(wfrag + f * 1024 + lane * 32);
        #pragma unroll
        for (int e = 0; e < 16; ++e) dst[e] = (bf16_t)wr[e];
    }

    // ---- h0 = initial + ih[:,0] for our column slice ----
    for (int e = tid; e < NB * JCH; e += 256) {
        const int n = e >> 6;
        const int j = j0 + (e & 63);
        const float v = initial[n * HH + j] + ihbuf[(size_t)(n * TSTEPS) * HH + j];
        hid[(size_t)(n * TSTEPS) * HH + j] = v;
        hbuf0[n * HH + j] = (bf16_t)v;
    }
    __syncthreads();
    grid_barrier(cnt, gen);

    const int mi  = wv & 3;            // wave's M tile (batch rows mi*16..+15)
    const int nib = (wv >> 2) * 2;     // wave's first N tile
    const unsigned lds_astage = (unsigned)(size_t)(void*)astage;

    for (int t = 1; t < TSTEPS; ++t) {
        const bf16_t* src = ((t - 1) & 1) ? hbuf1 : hbuf0;

        // ---- TDM: stage h_{t-1} (64 x 1024 bf16) into LDS, wave 0 only ----
        if (wv == 0) {
#if __has_builtin(__builtin_amdgcn_tensor_load_to_lds)
            unsigned long long ga = (unsigned long long)(size_t)src;
            u32x4 g0; i32x8 g1; i32x4 g2 = {0, 0, 0, 0}, g3 = {0, 0, 0, 0};
            i32x8 g4 = {0, 0, 0, 0, 0, 0, 0, 0};
            g0[0] = 1u;                                        // count=1
            g0[1] = lds_astage;                                // lds_addr
            g0[2] = (unsigned)ga;                              // global_addr[31:0]
            g0[3] = (unsigned)((ga >> 32) & 0x1FFFFFFu) | (2u << 30); // [56:32], type=2
            // data_size=2B, pad_enable, pad_interval=256dw(code7), pad_amount=4dw(code3)
            g1[0] = (int)((1u << 16) | (1u << 20) | (7u << 22) | (3u << 25));
            g1[1] = (int)(1024u << 16);   // tensor_dim0[15:0]=1024
            g1[2] = (int)(64u << 16);     // tensor_dim1[15:0]=64
            g1[3] = (int)(1024u << 16);   // tile_dim0=1024
            g1[4] = 64;                   // tile_dim1=64
            g1[5] = 1024;                 // tensor_dim0_stride=1024
            g1[6] = 0; g1[7] = 0;
            __builtin_amdgcn_tensor_load_to_lds(g0, g1, g2, g3, g4, 0);
            __builtin_amdgcn_s_wait_tensorcnt(0);
#else
            for (int e = lane; e < NB * HH / 8; e += 32) {     // fallback copy
                const int row = e >> 7, kq = (e & 127) * 8;
                u32x4 v = *(const u32x4*)(src + (size_t)row * HH + kq);
                *(u32x4*)(astage + row * AROW_BYTES + kq * 2 + (kq >= 512 ? 16 : 0)) = v;
            }
#endif
        }
        __syncthreads();

        // ---- 2 output tiles per wave, K-loop over 32 kb blocks ----
        v8f c0 = {}, c1 = {};
        #pragma unroll 4
        for (int kb = 0; kb < 32; ++kb) {
            const int fix = (kb >= 16) ? 16 : 0;   // TDM mid-row pad fixup
            const unsigned char* ap = astage + (mi * 16 + l15) * AROW_BYTES
                                      + (kb * 64 + lhalf * 32) + fix;
            pair16 ar = { *(const u32x4*)ap, *(const u32x4*)(ap + 16) };
            v16bf a = __builtin_bit_cast(v16bf, ar);

            const unsigned char* bp0 = wfrag + (nib * 32 + kb) * 1024 + lane * 32;
            const unsigned char* bp1 = bp0 + 32 * 1024;
            pair16 br0 = { *(const u32x4*)bp0, *(const u32x4*)(bp0 + 16) };
            pair16 br1 = { *(const u32x4*)bp1, *(const u32x4*)(bp1 + 16) };
            v16bf b0 = __builtin_bit_cast(v16bf, br0);
            v16bf b1 = __builtin_bit_cast(v16bf, br1);

            c0 = __builtin_amdgcn_wmma_f32_16x16x32_bf16(false, a, false, b0, (short)0, c0, false, false);
            c1 = __builtin_amdgcn_wmma_f32_16x16x32_bf16(false, a, false, b1, (short)0, c1, false, false);
        }

        // ---- epilogue: h_t = C + ih[:,t-1]; write fp32 hid + bf16 ping-pong ----
        bf16_t* dsth = (t & 1) ? hbuf1 : hbuf0;
        #pragma unroll
        for (int r = 0; r < 8; ++r) {
            const int m    = r + lhalf * 8;
            const int nbat = mi * 16 + m;
            const size_t bu = ((size_t)nbat * TSTEPS + (t - 1)) * HH;
            const size_t bo = ((size_t)nbat * TSTEPS + t) * HH;
            const int col0 = j0 + nib * 16 + l15;
            const int col1 = col0 + 16;
            const float v0 = c0[r] + ihbuf[bu + col0];
            const float v1 = c1[r] + ihbuf[bu + col1];
            hid[bo + col0] = v0;
            hid[bo + col1] = v1;
            dsth[nbat * HH + col0] = (bf16_t)v0;
            dsth[nbat * HH + col1] = (bf16_t)v1;
        }
        grid_barrier(cnt, gen);
    }
}

// ---------------------------------------------------------------------------
// Kernel 3: duplicate hiddens into second output copy (b128 streaming)
// ---------------------------------------------------------------------------
__global__ __launch_bounds__(256) void dup_out(const f32x4* __restrict__ s,
                                               f32x4* __restrict__ d, int n4) {
    int i = blockIdx.x * blockDim.x + threadIdx.x;
    const int stride = gridDim.x * blockDim.x;
    for (; i < n4; i += stride) d[i] = s[i];
}

// ---------------------------------------------------------------------------
extern "C" void kernel_launch(void* const* d_in, const int* in_sizes, int n_in,
                              void* d_out, int out_size, void* d_ws, size_t ws_size,
                              hipStream_t stream) {
    (void)in_sizes; (void)n_in; (void)out_size; (void)ws_size;
    const float* x       = (const float*)d_in[0];
    const float* initial = (const float*)d_in[1];
    const float* W_ih    = (const float*)d_in[2];
    const float* b_ih    = (const float*)d_in[3];
    const float* W_hh    = (const float*)d_in[4];

    float* out1 = (float*)d_out;                       // hiddens
    float* out2 = out1 + (size_t)NB * TSTEPS * HH;     // scratch for ih, then copy 2

    int*    bar   = (int*)d_ws;                        // [0]=cnt [1]=gen
    bf16_t* hbuf0 = (bf16_t*)((char*)d_ws + 256);      // 64*1024 bf16
    bf16_t* hbuf1 = hbuf0 + NB * HH;

    init_ws<<<1, 32, 0, stream>>>(bar);

    dim3 g1(HH / 64, (NB * TSTEPS) / 128);             // 16 x 512
    input_gemm<<<g1, 256, 0, stream>>>(x, W_ih, b_ih, out2);

    rnn_recur<<<GWG, 256, 0, stream>>>(initial, W_hh, out2, out1,
                                       hbuf0, hbuf1, bar, bar + 1);

    const int n4 = (int)((size_t)NB * TSTEPS * HH / 4);
    dup_out<<<4096, 256, 0, stream>>>((const f32x4*)out1, (f32x4*)out2, n4);
}
